// IPAdapterTimeImageProjection_83760452207085
// MI455X (gfx1250) — matrix operand order repaired and checked by reading, not compiled, not run on hardware
//
#include <hip/hip_runtime.h>
#include <math.h>

// ---------------------------------------------------------------------------
// IPAdapterTimeImageProjection forward for MI455X (gfx1250, wave32, WMMA).
// All large GEMMs use v_wmma_f32_16x16x32_bf16 with fp32 accumulation.
// bf16 A-tiles can be staged via the Tensor Data Mover (tensor_load_to_lds).
// ---------------------------------------------------------------------------

typedef __attribute__((ext_vector_type(16))) __bf16 v16bf;
typedef __attribute__((ext_vector_type(8)))  __bf16 v8bf;
typedef __attribute__((ext_vector_type(8)))  float  v8f;
typedef __attribute__((ext_vector_type(4)))  unsigned int v4u;
typedef __attribute__((ext_vector_type(8)))  int v8i;
typedef __attribute__((ext_vector_type(4)))  int v4i;

#define CB  32      // batch
#define CN  577     // image tokens
#define CE  1152
#define CH  1280
#define CO  2432
#define CD  4
#define CHD 64
#define CNH 20
#define CNQ 64
#define CTIN 320
#define CFF 5120
#define MX  (CB*CN)        // 18464 rows of h
#define MKV (CB*(CN+CNQ))  // 20512 rows of kv input
#define ML  (CB*CNQ)       // 2048 latent rows
#define KPAD 672           // 641 keys padded to multiple of 32

enum { A_BF16 = 0, A_F32 = 1, A_KV = 2 };
enum { B_WF32 = 0, B_NK16 = 1, B_KN16 = 2 };
enum { E_BF16 = 0, E_F32 = 1, E_BIAS = 2, E_BIAS_TEMB = 3, E_GELU = 4, E_RES = 5 };

struct GP {
    const void* A;  long lda, a0, aB, aH;
    const void* Bm; long ldb, b0, bB, bH;
    void*       C;  long ldc, c0, cB, cH;
    int M, N, Kd, hz;
    int nLimit, kLimit;
    const float* bias;
    const float* temb; int tembDiv;
    const float* res;
    const __bf16* hn; const __bf16* lm; const float* g; const float* bv;
};

// ---------------------------------------------------------------------------
// TDM: DMA a 64-row x 32-col bf16 tile (row stride lda elems) from global
// memory into LDS laid out with a 40-element row stride (pad 4 DWORDs after
// every 16 DWORDs). Rows past rowsValid read as zero (tensor OOB -> 0).
// Descriptor packing per CDNA5 ISA 8.3/8.4 (D# group0/group1).
// ---------------------------------------------------------------------------
__device__ inline void tdm_load_tile_a(const __bf16* gbase, long ldaElems,
                                       int rowsValid, void* ldsDst) {
    unsigned long long ga = (unsigned long long)(uintptr_t)gbase;
    unsigned int lds = (unsigned int)(uintptr_t)ldsDst;   // LDS offset in addr[31:0]
    v4u g0;
    g0[0] = 1u;                                        // count=1, no gather
    g0[1] = lds;                                       // lds_addr (bytes)
    g0[2] = (unsigned int)ga;                          // global_addr[31:0]
    g0[3] = (unsigned int)((ga >> 32) & 0x01FFFFFFu)   // global_addr[56:32]
            | (2u << 30);                              // type = 2 (image)
    unsigned long long s0 = (unsigned long long)ldaElems;  // dim0 stride (elems)
    unsigned int td0 = (unsigned int)ldaElems;             // tensor_dim0
    unsigned int td1 = (unsigned int)rowsValid;            // tensor_dim1
    v8i g1;
    g1[0] = (int)((1u << 16)        // data_size = 1 -> 2 bytes
                | (1u << 20)        // pad_enable
                | (3u << 22)        // pad_interval: 16 DWORDs
                | (3u << 25));      // pad_amount:   4 DWORDs
    g1[1] = (int)((td0 & 0xFFFFu) << 16);                    // tensor_dim0[15:0]
    g1[2] = (int)((td0 >> 16) | ((td1 & 0xFFFFu) << 16));    // dim0 hi | dim1 lo
    g1[3] = (int)((td1 >> 16) | (32u << 16));                // dim1 hi | tile_dim0=32
    g1[4] = 64;                                              // tile_dim1=64, tile_dim2=0
    g1[5] = (int)(unsigned int)s0;                           // dim0_stride[31:0]
    g1[6] = (int)(unsigned int)((s0 >> 32) & 0xFFFFu);       // dim0_stride[47:32]
    g1[7] = 0;
    v4i z4 = {0, 0, 0, 0};
#if __clang_major__ >= 23
    v8i z8 = {0, 0, 0, 0, 0, 0, 0, 0};
    __builtin_amdgcn_tensor_load_to_lds(g0, g1, z4, z4, z8, 0);
#else
    __builtin_amdgcn_tensor_load_to_lds(g0, g1, z4, z4, 0);
#endif
}

// ---------------------------------------------------------------------------
// Generic tiled WMMA GEMM: C[M,N] (+epilogue) = A[M,K] x B[K,N]
// Workgroup: 256 threads = 8 waves (2x4), tile 64(M) x 128(N), K step 32.
// ---------------------------------------------------------------------------
template <int AM, int BM, int EP, int TDMF>
__global__ __launch_bounds__(256) void gemm_k(GP p) {
    __shared__ __align__(16) __bf16 As[64][40];
    __shared__ __align__(16) __bf16 Bs[128][40];

    const int tid  = threadIdx.x;
    const int lane = tid & 31;
    const int wave = tid >> 5;
    const int wm   = (wave >> 2) * 32;   // 0 or 32
    const int wn   = (wave & 3) * 32;    // 0..96
    const int half = lane >> 4;
    const int l15  = lane & 15;
    const int m0   = blockIdx.y * 64;
    const int n0   = blockIdx.x * 128;

    int zb = 0, zh = 0;
    if (p.hz > 0) { zb = blockIdx.z / p.hz; zh = blockIdx.z % p.hz; }
    const long aoff = p.a0 + (long)zb * p.aB + (long)zh * p.aH;
    const long boff = p.b0 + (long)zb * p.bB + (long)zh * p.bH;
    const long coff = p.c0 + (long)zb * p.cB + (long)zh * p.cH;

    v8f acc[2][2] = {};

    for (int k0 = 0; k0 < p.Kd; k0 += 32) {
        // ---------------- stage A tile (64 x 32) into LDS as bf16 ----------
        if (TDMF && AM == A_BF16) {
            // Tensor Data Mover path: wave 0 DMAs the whole tile.
            if (wave == 0) {
                const __bf16* gbase =
                    (const __bf16*)p.A + aoff + (long)m0 * p.lda + k0;
                tdm_load_tile_a(gbase, p.lda, p.M - m0, &As[0][0]);
                __builtin_amdgcn_s_wait_tensorcnt(0);
            }
        } else {
            const int row = tid >> 2;
            const int seg = tid & 3;
            const int gm  = m0 + row;
            const int k   = k0 + seg * 8;
            v8bf v = {};
            if (gm < p.M) {
                if (AM == A_BF16) {
                    const __bf16* a = (const __bf16*)p.A + aoff + (long)gm * p.lda + k;
                    v = *(const v8bf*)a;
                } else if (AM == A_F32) {
                    const float* a = (const float*)p.A + aoff + (long)gm * p.lda + k;
                    #pragma unroll
                    for (int i = 0; i < 8; ++i) v[i] = (__bf16)a[i];
                } else { // A_KV: rows 0..576 of each batch from hn (+LN affine), 577..640 from lm
                    int b  = gm / 641;
                    int r2 = gm - 641 * b;
                    if (r2 < 577) {
                        const __bf16* src = p.hn + ((long)(b * 577 + r2)) * CH + k;
                        #pragma unroll
                        for (int i = 0; i < 8; ++i)
                            v[i] = (__bf16)((float)src[i] * p.g[k + i] + p.bv[k + i]);
                    } else {
                        const __bf16* src = p.lm + ((long)(b * CNQ + (r2 - 577))) * CH + k;
                        v = *(const v8bf*)src;
                    }
                }
            }
            *(v8bf*)&As[row][seg * 8] = v;
        }
        // ---------------- stage B tile (32 x 128) into LDS as bf16 ---------
        if (BM == B_WF32) {
            const int n  = (tid < 128) ? tid : tid - 128;
            const int kb = (tid < 128) ? 0 : 16;
            const int gn = n0 + n;
            const float* bp = (const float*)p.Bm + boff;
            if (k0 + 32 < p.Kd && gn < p.N)
                __builtin_prefetch(&bp[(long)(k0 + 32 + kb) * p.ldb + gn], 0, 1);
            #pragma unroll
            for (int i = 0; i < 16; ++i) {
                int kk = k0 + kb + i;
                float val = (gn < p.N) ? bp[(long)kk * p.ldb + gn] : 0.0f;
                Bs[n][kb + i] = (__bf16)val;
            }
        } else if (BM == B_KN16) {  // bf16 [k][n] source with k limit (V matrix)
            const int n  = (tid < 128) ? tid : tid - 128;
            const int kb = (tid < 128) ? 0 : 16;
            const int gn = n0 + n;
            const __bf16* bp = (const __bf16*)p.Bm + boff;
            #pragma unroll
            for (int i = 0; i < 16; ++i) {
                int kk = k0 + kb + i;
                __bf16 val = (__bf16)0.0f;
                if (gn < p.N && kk < p.kLimit) val = bp[(long)kk * p.ldb + gn];
                Bs[n][kb + i] = val;
            }
        } else {                    // B_NK16: bf16 [n][k] source (K matrix)
            const int n2 = tid >> 1;
            const int hs = tid & 1;
            const int gn = n0 + n2;
            v8bf v0 = {}, v1 = {};
            if (gn < p.nLimit) {
                const __bf16* bp = (const __bf16*)p.Bm + boff + (long)gn * p.ldb + k0 + hs * 16;
                v0 = *(const v8bf*)bp;
                v1 = *(const v8bf*)(bp + 8);
            }
            *(v8bf*)&Bs[n2][hs * 16]     = v0;
            *(v8bf*)&Bs[n2][hs * 16 + 8] = v1;
        }
        __syncthreads();

        // ---------------- fragments + WMMA ---------------------------------
        // A 16x32 bf16 layout: lanes0-15 (M=l15): K 0..7 (v0-3), 16..23 (v4-7)
        //                      lanes16-31: K 8..15, 24..31
        v16bf af[2], bfr[2];
        #pragma unroll
        for (int ti = 0; ti < 2; ++ti) {
            const int r = wm + ti * 16 + l15;
            v8bf lo = *(const v8bf*)&As[r][half * 8];
            v8bf hi = *(const v8bf*)&As[r][16 + half * 8];
            #pragma unroll
            for (int i = 0; i < 8; ++i) { af[ti][i] = lo[i]; af[ti][8 + i] = hi[i]; }
        }
        // B 32x16 bf16 layout: lanes0-15 (N=l15): K 0..7, 16..23 ; lanes16-31: K 8..15, 24..31
        #pragma unroll
        for (int tj = 0; tj < 2; ++tj) {
            const int r = wn + tj * 16 + l15;
            v8bf lo = *(const v8bf*)&Bs[r][half * 8];
            v8bf hi = *(const v8bf*)&Bs[r][16 + half * 8];
            #pragma unroll
            for (int i = 0; i < 8; ++i) { bfr[tj][i] = lo[i]; bfr[tj][8 + i] = hi[i]; }
        }
        #pragma unroll
        for (int ti = 0; ti < 2; ++ti)
            #pragma unroll
            for (int tj = 0; tj < 2; ++tj)
                acc[ti][tj] = __builtin_amdgcn_wmma_f32_16x16x32_bf16(
                    false, af[ti], false, bfr[tj], (short)0, acc[ti][tj], false, false);
        __syncthreads();
    }

    // ---------------- epilogue ---------------------------------------------
    #pragma unroll
    for (int ti = 0; ti < 2; ++ti)
        #pragma unroll
        for (int tj = 0; tj < 2; ++tj)
            #pragma unroll
            for (int pp = 0; pp < 8; ++pp) {
                int gm = m0 + wm + ti * 16 + pp + 8 * half;
                int gn = n0 + wn + tj * 16 + l15;
                if (gm >= p.M || gn >= p.N) continue;
                float v = acc[ti][tj][pp];
                long ci = coff + (long)gm * p.ldc + gn;
                if (EP == E_BF16) {
                    ((__bf16*)p.C)[ci] = (__bf16)v;
                } else if (EP == E_F32) {
                    ((float*)p.C)[ci] = v;
                } else if (EP == E_BIAS) {
                    ((float*)p.C)[ci] = v + p.bias[gn];
                } else if (EP == E_BIAS_TEMB) {
                    ((float*)p.C)[ci] = v + p.bias[gn] +
                                        p.temb[(long)(gm / p.tembDiv) * p.N + gn];
                } else if (EP == E_GELU) {
                    float gl = 0.5f * v * (1.0f + erff(v * 0.70710678118f));
                    ((__bf16*)p.C)[ci] = (__bf16)gl;
                } else { // E_RES
                    ((float*)p.C)[ci] = v + p.res[ci];
                }
            }
}

// ---------------------------------------------------------------------------
// Small-M (M=32) fused MLP kernel: C[32,N] = actOut(actIn(A)[32,K] @ W + bias)
// ---------------------------------------------------------------------------
template <bool SIN, bool SOUT>
__global__ __launch_bounds__(256) void rowmlp_k(const float* A, const float* W,
                                               const float* bias, float* C,
                                               int K, int N, long wz, long bz, long cz) {
    __shared__ float Asm[32][64];
    const int n = blockIdx.x * 256 + threadIdx.x;
    const float* Wp = W + (long)blockIdx.z * wz;
    const float* bp = bias + (long)blockIdx.z * bz;
    float* Cp = C + (long)blockIdx.z * cz;
    float acc[32];
    #pragma unroll
    for (int b = 0; b < 32; ++b) acc[b] = 0.f;
    for (int kc = 0; kc < K; kc += 64) {
        __syncthreads();
        for (int i = threadIdx.x; i < 32 * 64; i += 256) {
            int b = i >> 6, k = i & 63;
            float a = A[(long)b * K + kc + k];
            if (SIN) a = a / (1.f + expf(-a));
            Asm[b][k] = a;
        }
        __syncthreads();
        for (int k = 0; k < 64; ++k) {
            float w = (n < N) ? Wp[(long)(kc + k) * N + n] : 0.f;
            #pragma unroll
            for (int b = 0; b < 32; ++b) acc[b] += Asm[b][k] * w;
        }
    }
    if (n < N) {
        #pragma unroll
        for (int b = 0; b < 32; ++b) {
            float v = acc[b] + bp[n];
            if (SOUT) v = v / (1.f + expf(-v));
            Cp[(long)b * N + n] = v;
        }
    }
}

// ---------------------------------------------------------------------------
// Timestep sinusoidal embedding: [cos | sin], TIN=320
// ---------------------------------------------------------------------------
__global__ void timeproj_k(const float* t, float* out) {
    const int b = blockIdx.x, j = threadIdx.x;
    const int half = CTIN / 2;
    const int idx = (j < half) ? j : j - half;
    float freq = expf(-logf(10000.f) * (float)idx / (float)half);
    float ang = t[b] * freq;
    out[b * CTIN + j] = (j < half) ? cosf(ang) : sinf(ang);
}

// ---------------------------------------------------------------------------
// Reductions
// ---------------------------------------------------------------------------
__device__ inline float wave_sum(float v) {
    #pragma unroll
    for (int o = 16; o > 0; o >>= 1) v += __shfl_xor(v, o, 32);
    return v;
}

// LN stats over h (width 1280): hn = (x - mean)*rstd as bf16 (affine folded later)
__global__ __launch_bounds__(256) void lnstats_k(const float* x, __bf16* hn) {
    __shared__ float red[16];
    const long row = blockIdx.x;
    const float* xr = x + row * CH;
    float v[5], s = 0.f, sq = 0.f;
    #pragma unroll
    for (int i = 0; i < 5; ++i) {
        v[i] = xr[threadIdx.x + i * 256];
        s += v[i]; sq += v[i] * v[i];
    }
    s = wave_sum(s); sq = wave_sum(sq);
    const int wave = threadIdx.x >> 5, lane = threadIdx.x & 31;
    if (lane == 0) { red[wave] = s; red[8 + wave] = sq; }
    __syncthreads();
    s = 0.f; sq = 0.f;
    #pragma unroll
    for (int i = 0; i < 8; ++i) { s += red[i]; sq += red[8 + i]; }
    const float mean = s / (float)CH;
    const float rstd = rsqrtf(sq / (float)CH - mean * mean + 1e-5f);
    #pragma unroll
    for (int i = 0; i < 5; ++i)
        hn[row * CH + threadIdx.x + i * 256] = (__bf16)((v[i] - mean) * rstd);
}

// Adaptive LN of latents: out = (LN(x)*g+b)*(1+sc) + sh  -> bf16
__global__ __launch_bounds__(256) void adaln_k(const float* x, const float* g,
                                               const float* bb, const float* emb,
                                               int shOff, int scOff, __bf16* out) {
    __shared__ float red[16];
    const long row = blockIdx.x;               // 0..2047
    const int bidx = (int)(row >> 6);
    const float* xr = x + row * CH;
    const float* sh = emb + (long)bidx * (4 * CH) + shOff;
    const float* sc = emb + (long)bidx * (4 * CH) + scOff;
    float v[5], s = 0.f, sq = 0.f;
    #pragma unroll
    for (int i = 0; i < 5; ++i) {
        v[i] = xr[threadIdx.x + i * 256];
        s += v[i]; sq += v[i] * v[i];
    }
    s = wave_sum(s); sq = wave_sum(sq);
    const int wave = threadIdx.x >> 5, lane = threadIdx.x & 31;
    if (lane == 0) { red[wave] = s; red[8 + wave] = sq; }
    __syncthreads();
    s = 0.f; sq = 0.f;
    #pragma unroll
    for (int i = 0; i < 8; ++i) { s += red[i]; sq += red[8 + i]; }
    const float mean = s / (float)CH;
    const float rstd = rsqrtf(sq / (float)CH - mean * mean + 1e-5f);
    #pragma unroll
    for (int i = 0; i < 5; ++i) {
        int j = threadIdx.x + i * 256;
        float y = (v[i] - mean) * rstd * g[j] + bb[j];
        y = y * (1.f + sc[j]) + sh[j];
        out[row * CH + j] = (__bf16)y;
    }
}

// Softmax over 641 keys (scaled by 1/8), bf16 out, zero pad to 672. Wave/row.
__global__ __launch_bounds__(256) void softmax_k(const float* S, __bf16* P) {
    const long row = (long)blockIdx.x * 8 + (threadIdx.x >> 5);
    const int lane = threadIdx.x & 31;
    const float* sr = S + row * KPAD;
    float v[21];
    float m = -1e30f;
    #pragma unroll
    for (int i = 0; i < 21; ++i) {
        int j = lane + i * 32;
        v[i] = (j < 641) ? sr[j] * 0.125f : -1e30f;
        m = fmaxf(m, v[i]);
    }
    #pragma unroll
    for (int o = 16; o > 0; o >>= 1) m = fmaxf(m, __shfl_xor(m, o, 32));
    float s = 0.f;
    #pragma unroll
    for (int i = 0; i < 21; ++i) {
        int j = lane + i * 32;
        v[i] = (j < 641) ? expf(v[i] - m) : 0.f;
        s += v[i];
    }
    s = wave_sum(s);
    const float inv = 1.f / s;
    #pragma unroll
    for (int i = 0; i < 21; ++i) {
        int j = lane + i * 32;
        if (j < KPAD) P[row * KPAD + j] = (__bf16)(v[i] * inv);
    }
}

__global__ void latinit_k(const float* l0, float* lat) {
    long i = (long)blockIdx.x * 256 + threadIdx.x;
    if (i >= (long)ML * CH) return;
    long r = i / CH; int j = (int)(i - r * CH); int q = (int)(r & 63);
    lat[i] = l0[q * CH + j];
}

__global__ void castbf_k(const float* x, __bf16* y, long n) {
    long i = (long)blockIdx.x * 256 + threadIdx.x;
    if (i < n) y[i] = (__bf16)x[i];
}

__global__ __launch_bounds__(256) void lnfinal_k(const float* x, const float* g,
                                                 const float* bb, float* out) {
    __shared__ float red[16];
    const long row = blockIdx.x;
    const float* xr = x + row * CO;
    float s = 0.f, sq = 0.f;
    for (int j = threadIdx.x; j < CO; j += 256) { float a = xr[j]; s += a; sq += a * a; }
    s = wave_sum(s); sq = wave_sum(sq);
    const int wave = threadIdx.x >> 5, lane = threadIdx.x & 31;
    if (lane == 0) { red[wave] = s; red[8 + wave] = sq; }
    __syncthreads();
    s = 0.f; sq = 0.f;
    #pragma unroll
    for (int i = 0; i < 8; ++i) { s += red[i]; sq += red[8 + i]; }
    const float mean = s / (float)CO;
    const float rstd = rsqrtf(sq / (float)CO - mean * mean + 1e-5f);
    for (int j = threadIdx.x; j < CO; j += 256)
        out[row * CO + j] = (xr[j] - mean) * rstd * g[j] + bb[j];
}

__global__ void copy_k(const float* src, float* dst, long n) {
    long i = (long)blockIdx.x * 256 + threadIdx.x;
    if (i < n) dst[i] = src[i];
}

// ---------------------------------------------------------------------------
extern "C" void kernel_launch(void* const* d_in, const int* in_sizes, int n_in,
                              void* d_out, int out_size, void* d_ws, size_t ws_size,
                              hipStream_t stream) {
    const float* x         = (const float*)d_in[0];
    const float* timestep  = (const float*)d_in[1];
    const float* latents0  = (const float*)d_in[2];
    const float* proj_in_W = (const float*)d_in[3];
    const float* proj_in_b = (const float*)d_in[4];
    const float* te1_W     = (const float*)d_in[5];
    const float* te1_b     = (const float*)d_in[6];
    const float* te2_W     = (const float*)d_in[7];
    const float* te2_b     = (const float*)d_in[8];
    const float* ln0_g     = (const float*)d_in[9];
    const float* ln0_b     = (const float*)d_in[10];
    const float* ln1_g     = (const float*)d_in[11];
    const float* ln1_b     = (const float*)d_in[12];
    const float* Wq        = (const float*)d_in[13];
    const float* Wkv       = (const float*)d_in[14];
    const float* Wo        = (const float*)d_in[15];
    const float* ada_W     = (const float*)d_in[16];
    const float* ada_b     = (const float*)d_in[17];
    const float* lnada_g   = (const float*)d_in[18];
    const float* lnada_b   = (const float*)d_in[19];
    const float* Wff1      = (const float*)d_in[20];
    const float* Wff2      = (const float*)d_in[21];
    const float* proj_out_W= (const float*)d_in[22];
    const float* proj_out_b= (const float*)d_in[23];
    const float* norm_out_g= (const float*)d_in[24];
    const float* norm_out_b= (const float*)d_in[25];
    float* out = (float*)d_out;

    // workspace carving (256B aligned)
    char* w = (char*)d_ws;
    auto alloc = [&](size_t bytes) -> char* {
        char* p = w; w += (bytes + 255) & ~(size_t)255; return p;
    };
    float*  emb0   = (float*) alloc((size_t)CB * CTIN * 4);
    float*  hid    = (float*) alloc((size_t)CB * CH * 4);
    float*  temb   = (float*) alloc((size_t)CB * CH * 4);
    float*  embAll = (float*) alloc((size_t)CD * CB * 4 * CH * 4);
    float*  h      = (float*) alloc((size_t)MX * CH * 4);
    __bf16* hn     = (__bf16*)alloc((size_t)MX * CH * 2);
    float*  lat    = (float*) alloc((size_t)ML * CH * 4);
    __bf16* lmb    = (__bf16*)alloc((size_t)ML * CH * 2);
    __bf16* qb     = (__bf16*)alloc((size_t)ML * CH * 2);
    __bf16* attb   = (__bf16*)alloc((size_t)ML * CH * 2);
    __bf16* ffb    = (__bf16*)alloc((size_t)ML * CFF * 2);
    __bf16* kvb    = (__bf16*)alloc((size_t)MKV * 2 * CH * 2);
    float*  sbuf   = (float*) alloc((size_t)CB * CNH * CNQ * KPAD * 4);
    __bf16* pbuf   = (__bf16*)alloc((size_t)CB * CNH * CNQ * KPAD * 2);
    float*  obuf   = (float*) alloc((size_t)ML * CO * 4);

    // --- time embedding path ---
    timeproj_k<<<CB, CTIN, 0, stream>>>(timestep, emb0);
    rowmlp_k<false, true><<<dim3((CH + 255) / 256, 1, 1), 256, 0, stream>>>(
        emb0, te1_W, te1_b, hid, CTIN, CH, 0, 0, 0);
    rowmlp_k<false, false><<<dim3((CH + 255) / 256, 1, 1), 256, 0, stream>>>(
        hid, te2_W, te2_b, temb, CH, CH, 0, 0, 0);
    // per-layer adaLN embeddings, all layers at once (silu on input)
    rowmlp_k<true, false><<<dim3((4 * CH + 255) / 256, 1, CD), 256, 0, stream>>>(
        temb, ada_W, ada_b, embAll, CH, 4 * CH,
        (long)CH * 4 * CH, (long)4 * CH, (long)CB * 4 * CH);

    // --- latents init + proj_in + h LN stats ---
    latinit_k<<<(int)(((long)ML * CH + 255) / 256), 256, 0, stream>>>(latents0, lat);
    {
        GP p{}; p.A = x; p.lda = CE; p.Bm = proj_in_W; p.ldb = CH;
        p.C = h; p.ldc = CH; p.M = MX; p.N = CH; p.Kd = CE; p.hz = 1;
        p.bias = proj_in_b; p.temb = temb; p.tembDiv = CN;
        gemm_k<A_F32, B_WF32, E_BIAS_TEMB, 0>
            <<<dim3((CH + 127) / 128, (MX + 63) / 64, 1), 256, 0, stream>>>(p);
    }
    lnstats_k<<<MX, 256, 0, stream>>>(h, hn);

    // --- transformer layers ---
    for (int i = 0; i < CD; ++i) {
        const float* embL = embAll + (long)i * CB * 4 * CH;
        // lm = adaLN_msa(lat)
        adaln_k<<<ML, 256, 0, stream>>>(lat, ln1_g + (long)i * CH, ln1_b + (long)i * CH,
                                        embL, 0, CH, lmb);
        // q = lm @ Wq[i]   (A tile staged via Tensor Data Mover)
        {
            GP p{}; p.A = lmb; p.lda = CH; p.Bm = Wq; p.ldb = CH;
            p.b0 = (long)i * CH * CH; p.C = qb; p.ldc = CH;
            p.M = ML; p.N = CH; p.Kd = CH; p.hz = 1;
            gemm_k<A_BF16, B_WF32, E_BF16, 1>
                <<<dim3((CH + 127) / 128, (ML + 63) / 64, 1), 256, 0, stream>>>(p);
        }
        // kv = concat(LN(h)*g+b, lm) @ Wkv[i]   (LN affine folded into A loader)
        {
            GP p{}; p.hn = hn; p.lm = lmb;
            p.g = ln0_g + (long)i * CH; p.bv = ln0_b + (long)i * CH;
            p.Bm = Wkv; p.ldb = 2 * CH; p.b0 = (long)i * CH * 2 * CH;
            p.C = kvb; p.ldc = 2 * CH;
            p.M = MKV; p.N = 2 * CH; p.Kd = CH; p.hz = 1;
            gemm_k<A_KV, B_WF32, E_BF16, 0>
                <<<dim3((2 * CH + 127) / 128, (MKV + 63) / 64, 1), 256, 0, stream>>>(p);
        }
        // S = q @ k^T per (b,head)
        {
            GP p{}; p.A = qb; p.lda = CH; p.aB = (long)CNQ * CH; p.aH = CHD;
            p.Bm = kvb; p.ldb = 2 * CH; p.bB = (long)641 * 2 * CH; p.bH = CHD;
            p.nLimit = 641;
            p.C = sbuf; p.ldc = KPAD; p.cB = (long)CNH * CNQ * KPAD; p.cH = (long)CNQ * KPAD;
            p.M = CNQ; p.N = 641; p.Kd = CHD; p.hz = CNH;
            gemm_k<A_BF16, B_NK16, E_F32, 0>
                <<<dim3((641 + 127) / 128, 1, CB * CNH), 256, 0, stream>>>(p);
        }
        softmax_k<<<(CB * CNH * CNQ) / 8, 256, 0, stream>>>(sbuf, pbuf);
        // att = P @ V per (b,head)   (A tile staged via Tensor Data Mover)
        {
            GP p{}; p.A = pbuf; p.lda = KPAD;
            p.aB = (long)CNH * CNQ * KPAD; p.aH = (long)CNQ * KPAD;
            p.Bm = kvb; p.ldb = 2 * CH; p.b0 = CH;
            p.bB = (long)641 * 2 * CH; p.bH = CHD; p.kLimit = 641;
            p.C = attb; p.ldc = CH; p.cB = (long)CNQ * CH; p.cH = CHD;
            p.M = CNQ; p.N = CHD; p.Kd = KPAD; p.hz = CNH;
            gemm_k<A_BF16, B_KN16, E_BF16, 1>
                <<<dim3(1, 1, CB * CNH), 256, 0, stream>>>(p);
        }
        // lat = att @ Wo[i] + lat
        {
            GP p{}; p.A = attb; p.lda = CH; p.Bm = Wo; p.ldb = CH;
            p.b0 = (long)i * CH * CH; p.C = lat; p.ldc = CH; p.res = lat;
            p.M = ML; p.N = CH; p.Kd = CH; p.hz = 1;
            gemm_k<A_BF16, B_WF32, E_RES, 0>
                <<<dim3((CH + 127) / 128, (ML + 63) / 64, 1), 256, 0, stream>>>(p);
        }
        // lm2 = adaLN_mlp(lat)
        adaln_k<<<ML, 256, 0, stream>>>(lat, lnada_g + (long)i * CH, lnada_b + (long)i * CH,
                                        embL, 2 * CH, 3 * CH, lmb);
        // ff = gelu(lm2 @ Wff1[i])
        {
            GP p{}; p.A = lmb; p.lda = CH; p.Bm = Wff1; p.ldb = CFF;
            p.b0 = (long)i * CH * CFF; p.C = ffb; p.ldc = CFF;
            p.M = ML; p.N = CFF; p.Kd = CH; p.hz = 1;
            gemm_k<A_BF16, B_WF32, E_GELU, 0>
                <<<dim3((CFF + 127) / 128, (ML + 63) / 64, 1), 256, 0, stream>>>(p);
        }
        // lat = ff @ Wff2[i] + lat
        {
            GP p{}; p.A = ffb; p.lda = CFF; p.Bm = Wff2; p.ldb = CH;
            p.b0 = (long)i * CFF * CH; p.C = lat; p.ldc = CH; p.res = lat;
            p.M = ML; p.N = CH; p.Kd = CFF; p.hz = 1;
            gemm_k<A_BF16, B_WF32, E_RES, 0>
                <<<dim3((CH + 127) / 128, (ML + 63) / 64, 1), 256, 0, stream>>>(p);
        }
    }

    // --- output projection + final LN; tuple tail = t_emb ---
    castbf_k<<<(int)(((long)ML * CH + 255) / 256), 256, 0, stream>>>(lat, lmb, (long)ML * CH);
    {
        GP p{}; p.A = lmb; p.lda = CH; p.Bm = proj_out_W; p.ldb = CO;
        p.C = obuf; p.ldc = CO; p.M = ML; p.N = CO; p.Kd = CH; p.hz = 1;
        p.bias = proj_out_b;
        gemm_k<A_BF16, B_WF32, E_BIAS, 0>
            <<<dim3((CO + 127) / 128, (ML + 63) / 64, 1), 256, 0, stream>>>(p);
    }
    lnfinal_k<<<ML, 256, 0, stream>>>(obuf, norm_out_g, norm_out_b, out);
    copy_k<<<((CB * CH) + 255) / 256, 256, 0, stream>>>(temb, out + (long)ML * CO,
                                                        (long)CB * CH);
}
